// SelfAtt_Spatial_24094766530773
// MI455X (gfx1250) — compile-verified, matrix operand order checked by reading, not run on hardware
//
#include <hip/hip_runtime.h>
#include <hip/hip_bf16.h>
#include <cstdint>
#include <cstddef>

// Problem constants (from reference)
#define NN    16
#define CC    256
#define COO   32
#define HWW   3200          // H*W = 25*128
#define NTI   200           // HWW / 16
#define BN_EPS 1e-5f

typedef _Float16 v8h  __attribute__((ext_vector_type(8)));
typedef _Float16 v16h __attribute__((ext_vector_type(16)));
typedef float    v8f  __attribute__((ext_vector_type(8)));

// Assemble one 16-bit WMMA operand fragment for K=32:
// elems 0..7  = row[halfK + 0..7], elems 8..15 = row[halfK + 16 .. halfK + 23]
static __device__ __forceinline__ v16h load_a32(const _Float16* base) {
  v8h lo = *(const v8h*)(base);
  v8h hi = *(const v8h*)(base + 16);
  v16h r;
#pragma unroll
  for (int e = 0; e < 8; ++e) { r[e] = lo[e]; r[e + 8] = hi[e]; }
  return r;
}

static __device__ __forceinline__ v8f wmma_f16(v16h a, v16h b, v8f c) {
  return __builtin_amdgcn_wmma_f32_16x16x32_f16(false, a, false, b, (short)0, c, false, false);
}

// ---------------------------------------------------------------------------
// Kernel 1: x (n,c,hw) f32 -> xh (n,hw,c) f16
// ---------------------------------------------------------------------------
__global__ void k_x_to_f16(const float* __restrict__ x, _Float16* __restrict__ xh) {
  int idx = blockIdx.x * blockDim.x + threadIdx.x;
  if (idx >= NN * CC * HWW) return;
  int i = idx % HWW;
  int c = (idx / HWW) % CC;
  int n = idx / (CC * HWW);
  xh[((size_t)n * HWW + i) * CC + c] = (_Float16)x[idx];
}

// ---------------------------------------------------------------------------
// Kernel 2: weight conversion f32 -> f16 (layouts already row-major as needed)
// ---------------------------------------------------------------------------
__global__ void k_w_to_f16(const float* __restrict__ wq, const float* __restrict__ wk,
                           const float* __restrict__ wv, const float* __restrict__ wo,
                           _Float16* __restrict__ wq16, _Float16* __restrict__ wk16,
                           _Float16* __restrict__ wv16, _Float16* __restrict__ wo16) {
  int idx = blockIdx.x * blockDim.x + threadIdx.x;
  if (idx < COO * CC) {
    wq16[idx] = (_Float16)wq[idx];
    wk16[idx] = (_Float16)wk[idx];
    wv16[idx] = (_Float16)wv[idx];
  }
  if (idx < CC * COO) wo16[idx] = (_Float16)wo[idx];
}

// ---------------------------------------------------------------------------
// Kernel 3: QKV projection. One wave per (i-tile, n).
// q,k -> (n,i,co) f16 ; v -> (n,co,i) f16
// ---------------------------------------------------------------------------
__global__ void __launch_bounds__(32) k_qkv(
    const _Float16* __restrict__ xh,
    const _Float16* __restrict__ wq16, const _Float16* __restrict__ wk16,
    const _Float16* __restrict__ wv16,
    const float* __restrict__ bq, const float* __restrict__ bk, const float* __restrict__ bv,
    _Float16* __restrict__ qh, _Float16* __restrict__ kh, _Float16* __restrict__ vh) {
  const int it = blockIdx.x;
  const int n  = blockIdx.y;
  const int lane  = threadIdx.x;
  const int l16   = lane & 15;
  const int halfK = (lane >> 4) * 8;
  const int i0 = it * 16;

  v8f accq0 = {}, accq1 = {}, acck0 = {}, acck1 = {}, accv0 = {}, accv1 = {};
  const _Float16* arow = xh + ((size_t)n * HWW + (i0 + l16)) * CC + halfK;

  for (int ks = 0; ks < 8; ++ks) {             // K = 256 = 8 x 32
    v16h a = load_a32(arow + ks * 32);
    const size_t wb0 = (size_t)l16 * CC + ks * 32 + halfK;
    const size_t wb1 = (size_t)(16 + l16) * CC + ks * 32 + halfK;
    accq0 = wmma_f16(a, load_a32(wq16 + wb0), accq0);
    accq1 = wmma_f16(a, load_a32(wq16 + wb1), accq1);
    acck0 = wmma_f16(a, load_a32(wk16 + wb0), acck0);
    acck1 = wmma_f16(a, load_a32(wk16 + wb1), acck1);
    accv0 = wmma_f16(a, load_a32(wv16 + wb0), accv0);
    accv1 = wmma_f16(a, load_a32(wv16 + wb1), accv1);
  }

#pragma unroll
  for (int t = 0; t < 2; ++t) {
    const int co = t * 16 + l16;
    const float biq = bq[co], bik = bk[co], biv = bv[co];
    const v8f aq = t ? accq1 : accq0;
    const v8f ak = t ? acck1 : acck0;
    const v8f av = t ? accv1 : accv0;
#pragma unroll
    for (int r = 0; r < 8; ++r) {
      const int i = i0 + r + 8 * (lane >> 4);
      const size_t rq = ((size_t)n * HWW + i) * COO + co;
      qh[rq] = (_Float16)(aq[r] + biq);
      kh[rq] = (_Float16)(ak[r] + bik);
      vh[((size_t)n * COO + co) * HWW + i] = (_Float16)(av[r] + biv);
    }
  }
}

// ---------------------------------------------------------------------------
// Kernel 4: fused energy -> softmax -> attention store -> P @ V^T
// One workgroup (10 waves) per (i-tile, n). 16x3200 energy strip lives in LDS.
// Dynamic LDS: 51200 f32 energy + 5120 f32 scratch = 225280 bytes (< 320 KB).
// ---------------------------------------------------------------------------
#define ATT_WAVES    10
#define ATT_THREADS  320
#define JT_PER_WAVE  20
#define LDS_E_FLOATS (16 * HWW)
#define LDS_SCR_FLOATS (ATT_WAVES * 512)
#define ATT_SMEM_BYTES ((LDS_E_FLOATS + LDS_SCR_FLOATS) * 4)

__global__ void __launch_bounds__(ATT_THREADS) k_attention(
    const _Float16* __restrict__ qh, const _Float16* __restrict__ kh,
    const _Float16* __restrict__ vh,
    float* __restrict__ attn_out,        // (n, HW, HW) f32, part of d_out
    _Float16* __restrict__ attc16) {     // (n, HW, CO) f16
  extern __shared__ float sm[];
  float* eLDS = sm;                       // [16][3200]
  float* scr  = sm + LDS_E_FLOATS;        // 5120 f32

  const int it  = blockIdx.x;
  const int n   = blockIdx.y;
  const int tid = threadIdx.x;
  const int w    = tid >> 5;
  const int lane = tid & 31;
  const int l16   = lane & 15;
  const int halfK = (lane >> 4) * 8;
  const int i0  = it * 16;
  const int jt0 = w * JT_PER_WAVE;

  // ---- phase 1: energy tiles (one WMMA each, K=32 done in a single step) ----
  const v16h aq = load_a32(qh + ((size_t)n * HWW + i0 + l16) * COO + halfK);
  for (int jj = 0; jj < JT_PER_WAVE; ++jj) {
    const int j0 = (jt0 + jj) * 16;
    const _Float16* kb = kh + ((size_t)n * HWW + j0 + l16) * COO + halfK;
    if (jj + 1 < JT_PER_WAVE) __builtin_prefetch(kb + 16 * COO, 0, 1);
    v8f e = {};
    e = wmma_f16(aq, load_a32(kb), e);
#pragma unroll
    for (int r = 0; r < 8; ++r) {
      const int m = r + 8 * (lane >> 4);
      eLDS[m * HWW + j0 + l16] = e[r];
    }
  }
  __syncthreads();

  // ---- phase 2: row-wise max and sum(exp) ----
  {
    const int row = tid & 15;
    const int seg = tid >> 4;                       // 0..19
    const float* er = eLDS + row * HWW + seg * 160; // 3200 / 20 = 160
    float mx = -3.0e38f;
    for (int q = 0; q < 160; ++q) mx = fmaxf(mx, er[q]);
    scr[tid] = mx;
    __syncthreads();
    if (tid < 16) {
      float m2 = -3.0e38f;
      for (int s = 0; s < 20; ++s) m2 = fmaxf(m2, scr[s * 16 + tid]);
      scr[ATT_THREADS + tid] = m2;                  // rowMax
    }
    __syncthreads();
    const float rowM = scr[ATT_THREADS + row];
    float sum = 0.f;
    for (int q = 0; q < 160; ++q) sum += __expf(er[q] - rowM);
    scr[tid] = sum;
    __syncthreads();
    if (tid < 16) {
      float s2 = 0.f;
      for (int s = 0; s < 20; ++s) s2 += scr[s * 16 + tid];
      scr[ATT_THREADS + 16 + tid] = 1.0f / s2;      // rowInv
    }
    __syncthreads();
  }

  // ---- phase 3: probabilities -> global attention + back into LDS ----
  {
    const float* rowMax = scr + ATT_THREADS;
    const float* rowInv = scr + ATT_THREADS + 16;
    float* outp = attn_out + ((size_t)n * HWW + i0) * HWW;
    for (int jj = 0; jj < JT_PER_WAVE; ++jj) {
      const int j0 = (jt0 + jj) * 16;
#pragma unroll
      for (int r = 0; r < 8; ++r) {
        const int m = r + 8 * (lane >> 4);
        const int col = j0 + l16;
        const float p = __expf(eLDS[m * HWW + col] - rowMax[m]) * rowInv[m];
        eLDS[m * HWW + col] = p;
        outp[(size_t)m * HWW + col] = p;
      }
    }
  }
  __syncthreads();

  // ---- phase 4: att2 = P @ V^T over this wave's 320 j-columns (K=32 pairs) ----
  v8f acc0 = {}, acc1 = {};
  for (int ks = 0; ks < JT_PER_WAVE / 2; ++ks) {
    const int jkk = jt0 * 16 + ks * 32;
    const float* pr = eLDS + l16 * HWW + jkk + halfK;
    v16h a;
#pragma unroll
    for (int e = 0; e < 8; ++e) { a[e] = (_Float16)pr[e]; a[e + 8] = (_Float16)pr[16 + e]; }
    const _Float16* vb0 = vh + ((size_t)n * COO + l16) * HWW + jkk + halfK;
    const _Float16* vb1 = vh + ((size_t)n * COO + 16 + l16) * HWW + jkk + halfK;
    acc0 = wmma_f16(a, load_a32(vb0), acc0);
    acc1 = wmma_f16(a, load_a32(vb1), acc1);
  }
#pragma unroll
  for (int r = 0; r < 8; ++r) {
    const int m = r + 8 * (lane >> 4);
    scr[w * 512 + m * 32 + l16]      = acc0[r];
    scr[w * 512 + m * 32 + 16 + l16] = acc1[r];
  }
  __syncthreads();
  for (int idx = tid; idx < 512; idx += ATT_THREADS) {
    float s = 0.f;
#pragma unroll
    for (int wv = 0; wv < ATT_WAVES; ++wv) s += scr[wv * 512 + idx];
    const int m = idx >> 5, co = idx & 31;
    attc16[((size_t)n * HWW + i0 + m) * COO + co] = (_Float16)s;
  }
}

// ---------------------------------------------------------------------------
// Kernel 5: output projection (51200 x 32) @ (32 x 256), per-channel stats.
// One wave per 16-row tile; 16 column tiles of 16 channels (single WMMA each).
// ---------------------------------------------------------------------------
__global__ void __launch_bounds__(32) k_proj(
    const _Float16* __restrict__ attc16, const _Float16* __restrict__ wo16,
    const float* __restrict__ bo,
    float* __restrict__ attf, float* __restrict__ sums) {
  const int wg   = blockIdx.x;                 // 0 .. 3199
  const int lane = threadIdx.x;
  const int l16   = lane & 15;
  const int halfK = (lane >> 4) * 8;
  const int g0  = wg * 16;
  const int n   = g0 / HWW;
  const int i0g = g0 % HWW;

  const v16h a = load_a32(attc16 + ((size_t)g0 + l16) * COO + halfK);
  float* sumP  = sums;
  float* sumSq = sums + CC;

  for (int ct = 0; ct < 16; ++ct) {
    const int cout = ct * 16 + l16;
    v8f d = {};
    d = wmma_f16(a, load_a32(wo16 + (size_t)cout * COO + halfK), d);
    const float bias = bo[cout];
    float ls = 0.f, lq = 0.f;
#pragma unroll
    for (int r = 0; r < 8; ++r) {
      const int i = i0g + r + 8 * (lane >> 4);
      const float v = d[r] + bias;
      attf[((size_t)n * CC + cout) * HWW + i] = v;
      ls += v;
      lq += v * v;
    }
    atomicAdd(&sumP[cout], ls);
    atomicAdd(&sumSq[cout], lq);
  }
}

// ---------------------------------------------------------------------------
// Kernel 6: batch-norm finalize + gamma * att + x
// ---------------------------------------------------------------------------
__global__ void k_bn(const float* __restrict__ attf, const float* __restrict__ x,
                     const float* __restrict__ sums,
                     const float* __restrict__ bn_w, const float* __restrict__ bn_b,
                     const float* __restrict__ gamma, float* __restrict__ out) {
  int idx = blockIdx.x * blockDim.x + threadIdx.x;
  if (idx >= NN * CC * HWW) return;
  const int c = (idx / HWW) % CC;
  const float inv = 1.0f / (float)(NN * HWW);
  const float mean = sums[c] * inv;
  const float var  = sums[CC + c] * inv - mean * mean;
  const float a = (attf[idx] - mean) * rsqrtf(var + BN_EPS);
  out[idx] = gamma[0] * (a * bn_w[c] + bn_b[c]) + x[idx];
}

// ---------------------------------------------------------------------------
extern "C" void kernel_launch(void* const* d_in, const int* in_sizes, int n_in,
                              void* d_out, int out_size, void* d_ws, size_t ws_size,
                              hipStream_t stream) {
  (void)in_sizes; (void)n_in; (void)out_size; (void)ws_size;
  const float* x     = (const float*)d_in[0];
  const float* wq    = (const float*)d_in[1];
  const float* bq    = (const float*)d_in[2];
  const float* wk    = (const float*)d_in[3];
  const float* bk    = (const float*)d_in[4];
  const float* wv    = (const float*)d_in[5];
  const float* bv    = (const float*)d_in[6];
  const float* wo    = (const float*)d_in[7];
  const float* bo    = (const float*)d_in[8];
  const float* bn_w  = (const float*)d_in[9];
  const float* bn_b  = (const float*)d_in[10];
  const float* gamma = (const float*)d_in[11];

  float* out_f  = (float*)d_out;                       // (n, c, h, w) f32
  float* attn_o = out_f + (size_t)NN * CC * HWW;       // (n, HW, HW) f32

  // workspace layout (all 256-byte aligned by construction)
  char* ws = (char*)d_ws;
  size_t off = 0;
  _Float16* xh    = (_Float16*)(ws + off); off += (size_t)NN * HWW * CC * 2;   // 26,214,400
  _Float16* wq16  = (_Float16*)(ws + off); off += (size_t)COO * CC * 2;
  _Float16* wk16  = (_Float16*)(ws + off); off += (size_t)COO * CC * 2;
  _Float16* wv16  = (_Float16*)(ws + off); off += (size_t)COO * CC * 2;
  _Float16* wo16  = (_Float16*)(ws + off); off += (size_t)CC * COO * 2;
  _Float16* qh    = (_Float16*)(ws + off); off += (size_t)NN * HWW * COO * 2;
  _Float16* kh    = (_Float16*)(ws + off); off += (size_t)NN * HWW * COO * 2;
  _Float16* vh    = (_Float16*)(ws + off); off += (size_t)NN * HWW * COO * 2;
  _Float16* attc  = (_Float16*)(ws + off); off += (size_t)NN * HWW * COO * 2;
  float*    attf  = (float*)(ws + off);    off += (size_t)NN * CC * HWW * 4;   // 52,428,800
  float*    sums  = (float*)(ws + off);    off += 2 * CC * 4;

  // 1) layout conversions
  k_x_to_f16<<<(NN * CC * HWW + 255) / 256, 256, 0, stream>>>(x, xh);
  k_w_to_f16<<<(COO * CC + 255) / 256, 256, 0, stream>>>(wq, wk, wv, wo, wq16, wk16, wv16, wo16);

  // 2) QKV projections (WMMA, K=256 in 8 steps)
  k_qkv<<<dim3(NTI, NN), 32, 0, stream>>>(xh, wq16, wk16, wv16, bq, bk, bv, qh, kh, vh);

  // 3) fused energy/softmax/attention/PV (WMMA + 225 KB LDS strip)
  k_attention<<<dim3(NTI, NN), ATT_THREADS, ATT_SMEM_BYTES, stream>>>(qh, kh, vh, attn_o, attc);

  // 4) output projection + per-channel sums (zero the accumulators first)
  hipMemsetAsync(sums, 0, 2 * CC * sizeof(float), stream);
  k_proj<<<(NN * HWW) / 16, 32, 0, stream>>>(attc, wo16, bo, attf, sums);

  // 5) batch-norm finalize + residual
  k_bn<<<(NN * CC * HWW + 255) / 256, 256, 0, stream>>>(attf, x, sums, bn_w, bn_b, gamma, out_f);
}
